// GATLayer_43885975830916
// MI455X (gfx1250) — matrix operand order repaired and checked by reading, not compiled, not run on hardware
//
#include <hip/hip_runtime.h>

// ---------------------------------------------------------------------------
// GAT layer fused flash-attention for MI455X (gfx1250, wave32, WMMA + async
// global->LDS pipeline).  B=2, N=2048, C=128, H=16 heads, c=8.
// ---------------------------------------------------------------------------

typedef __attribute__((ext_vector_type(16))) _Float16 v16h;
typedef __attribute__((ext_vector_type(8)))  _Float16 v8h;
typedef __attribute__((ext_vector_type(8)))  float    v8f;
typedef __attribute__((ext_vector_type(4)))  float    v4f;

#define NN   2048
#define BB   2
#define CC   128
#define NEG_BIG (-9e15f)

// LDS strides chosen for conflict-free ds_load_b128 (see analysis)
#define HSTRIDE 40   // f16 elems per hT-slab row (80B = 5 x 16B)
#define ASTRIDE 36   // f32 elems per adj row    (144B = 9 x 16B)
#define LSTRIDE 32   // f32 elems per lc row     (broadcast reads)

// ---- CDNA5 async global->LDS (guarded; falls back to load+ds_store) -------
#if defined(__has_builtin)
# if __has_builtin(__builtin_amdgcn_global_load_async_to_lds_b128)
#  define USE_ASYNC 1
# endif
#endif
#ifndef USE_ASYNC
# define USE_ASYNC 0
#endif

// builtin signature: void(v4i addrspace(1)*, v4i addrspace(3)*, Imm off, Imm cpol)
typedef int v4i __attribute__((vector_size(16)));
typedef __attribute__((address_space(1))) v4i gv4i;
typedef __attribute__((address_space(3))) v4i lv4i;

__device__ __forceinline__ void cp16_async(void* lds, const void* g) {
#if USE_ASYNC
    __builtin_amdgcn_global_load_async_to_lds_b128((gv4i*)g, (lv4i*)lds, 0, 0);
#else
    *(v8h*)lds = *(const v8h*)g;   // sync fallback (16B)
#endif
}

__device__ __forceinline__ void wait_async() {
#if USE_ASYNC
# if __has_builtin(__builtin_amdgcn_s_wait_asynccnt)
    __builtin_amdgcn_s_wait_asynccnt(0);
# else
    asm volatile("s_wait_asynccnt 0x0" ::: "memory");
# endif
#endif
}

// ===========================================================================
// Kernel 1: h = X @ W + b via v_wmma_f32_16x16x32_f16 (f16 in / f32 acc).
// Writes h TRANSPOSED  hT[b][feat][n] (f16)  and logits lpT/lcT[b][head][n]
// (f32, transposed so kernel 2 can stage them with contiguous async copies).
// One wave = one 16x16 tile; 8 waves/block; 256 blocks.
// ===========================================================================
__global__ __launch_bounds__(256) void gat_proj_kernel(
    const float* __restrict__ X,     // [B*N, 128]
    const float* __restrict__ W,     // [128, 128]
    const float* __restrict__ bias,  // [128]
    const float* __restrict__ aw,    // [16, 16]
    _Float16*    __restrict__ hT,    // [B][128][N]
    float*       __restrict__ lpT,   // [B][16][N]
    float*       __restrict__ lcT)   // [B][16][N]
{
    const int lane = threadIdx.x & 31;
    const int wave = threadIdx.x >> 5;
    const int tile = blockIdx.x * 8 + wave;    // 0..2047
    const int rt   = tile >> 3;
    const int ct   = tile & 7;
    const int n0   = rt * 16;                  // flat row base (b*N+n)
    const int l15  = lane & 15;
    const int lh   = lane >> 4;

    const int nrow = n0 + l15;
    const int col  = ct * 16 + l15;

    v8f acc = {};

    #pragma unroll
    for (int k0 = 0; k0 < CC; k0 += 32) {
        // A: X tile [16 n x 32 k].  lane<16: K 0..7 | 16..23 ; lane>=16: +8
        const float* xp = X + nrow * CC + k0 + lh * 8;
        v4f x0 = *(const v4f*)(xp);
        v4f x1 = *(const v4f*)(xp + 4);
        v4f x2 = *(const v4f*)(xp + 16);
        v4f x3 = *(const v4f*)(xp + 20);
        v16h afrag;
        #pragma unroll
        for (int e = 0; e < 4; ++e) {
            afrag[e]      = (_Float16)x0[e];
            afrag[e + 4]  = (_Float16)x1[e];
            afrag[e + 8]  = (_Float16)x2[e];
            afrag[e + 12] = (_Float16)x3[e];
        }
        // B: W tile [32 k x 16 feats], lane = column, elems = K 0..7 | 16..23 ; lane>=16: +8
        const float* wp = W + (k0 + lh * 8) * CC + col;
        v16h bfrag;
        #pragma unroll
        for (int e = 0; e < 16; ++e)
            bfrag[e] = (_Float16)wp[(e + (e & 8)) * CC];

        acc = __builtin_amdgcn_wmma_f32_16x16x32_f16(
            false, afrag, false, bfrag, (short)0, acc, false, false);
    }

    const float bv = bias[col];
    #pragma unroll
    for (int r = 0; r < 8; ++r) acc[r] += bv;

    // store hT: lane holds feat=col, n = n0 + lh*8 + r  (8 contiguous f16)
    const int b    = n0 >> 11;
    const int nloc = (n0 & (NN - 1)) + lh * 8;
    v8h hh;
    #pragma unroll
    for (int r = 0; r < 8; ++r) hh[r] = (_Float16)acc[r];
    *(v8h*)(hT + ((size_t)b * CC + col) * NN + nloc) = hh;

    // lp/lc: reduce the 8 feats of a head across lanes (fp32 exact)
    const int head = col >> 3;
    const int cw   = lane & 7;
    const float awl = aw[head * 16 + cw];
    const float awc = aw[head * 16 + 8 + cw];
    #pragma unroll
    for (int r = 0; r < 8; ++r) {
        float vl = acc[r] * awl;
        float vc = acc[r] * awc;
        vl += __shfl_xor(vl, 1, 32); vl += __shfl_xor(vl, 2, 32); vl += __shfl_xor(vl, 4, 32);
        vc += __shfl_xor(vc, 1, 32); vc += __shfl_xor(vc, 2, 32); vc += __shfl_xor(vc, 4, 32);
        if ((lane & 7) == 0) {
            const int nn = nloc + r;
            lpT[((size_t)b * 16 + head) * NN + nn] = vl;
            lcT[((size_t)b * 16 + head) * NN + nn] = vc;
        }
    }
}

// ===========================================================================
// Kernel 2: fused flash-GAT attention, 32-j tiles, double-buffered LDS,
// async global->LDS staging, 1 barrier/iter.  One lane owns one (i, head):
// online-softmax state and the 8-float O column are purely per-lane.
// Per wave per 16-j subtile: one v_wmma_f32_16x16x32_f16
//   D(O^T) = A(blockdiag[Vh0^T|Vh1^T]) x B([Ph0^T;Ph1^T]) + C
// ===========================================================================
__global__ __launch_bounds__(256) void gat_attn_kernel(
    const float*    __restrict__ adj,  // [B, N, N]
    const _Float16* __restrict__ hT,   // [B][128][N]
    const float*    __restrict__ lpT,  // [B][16][N]
    const float*    __restrict__ lcT,  // [B][16][N]
    float*          __restrict__ out)  // [B*N, 128]
{
    __shared__ _Float16 hTs[2][CC * HSTRIDE];   // [feat][32 j], 80B rows
    __shared__ float    adjS[2][16 * ASTRIDE];  // [i][32 j], 144B rows
    __shared__ float    lcS[2][16 * LSTRIDE];   // [head][32 j]

    const int tid  = threadIdx.x;
    const int lane = tid & 31;
    const int hp   = tid >> 5;            // head pair 0..7
    const int b    = blockIdx.x >> 7;
    const int i0   = (blockIdx.x & 127) * 16;
    const int l15  = lane & 15;
    const int lh   = lane >> 4;
    const int head = hp * 2 + lh;
    const int nf   = b * NN + i0 + l15;

    const float     lpv = lpT[((size_t)b * 16 + head) * NN + i0 + l15];
    const _Float16* hTb = hT + (size_t)b * CC * NN;
    const float*    lcb = lcT + (size_t)b * 16 * NN;

    // cooperative stage of one 32-j tile into buffer `dst`
    auto stage = [&](int j0, int dst) {
        // hT slab: 128 feats x 64B  -> 512 x 16B chunks, 2 per thread
        {
            int c = tid, feat = c >> 2, q = c & 3;
            cp16_async(&hTs[dst][feat * HSTRIDE + q * 8],
                       hTb + (size_t)feat * NN + j0 + q * 8);
            c = tid + 256; feat = c >> 2; q = c & 3;
            cp16_async(&hTs[dst][feat * HSTRIDE + q * 8],
                       hTb + (size_t)feat * NN + j0 + q * 8);
        }
        if (tid < 128) {   // adj tile: 16 rows x 128B (waves 0-3)
            int i = tid >> 3, q = tid & 7;
            cp16_async(&adjS[dst][i * ASTRIDE + q * 4],
                       adj + ((size_t)b * NN + i0 + i) * NN + j0 + q * 4);
        } else {           // lc tile: 16 heads x 128B (waves 4-7)
            int c = tid - 128, hd = c >> 3, q = c & 7;
            cp16_async(&lcS[dst][hd * LSTRIDE + q * 4],
                       lcb + (size_t)hd * NN + j0 + q * 4);
        }
    };

    float m = -__builtin_inff();
    float s = 0.0f;
    v8f acc = {};

    stage(0, 0);
    wait_async();
    __syncthreads();

    for (int jt = 0; jt < NN / 32; ++jt) {
        const int cur = jt & 1;
        if (jt + 1 < NN / 32) stage((jt + 1) * 32, cur ^ 1);   // prefetch

        const _Float16* hcur = &hTs[cur][0];
        const float*    acur = &adjS[cur][0];
        const float*    lcur = &lcS[cur][0];

        #pragma unroll
        for (int js = 0; js < 2; ++js) {
            const v4f* lrow = (const v4f*)(lcur + head * LSTRIDE + js * 16);
            const v4f* arow = (const v4f*)(acur + l15 * ASTRIDE + js * 16);
            float pl[16];
            float tmax = NEG_BIG;
            #pragma unroll
            for (int q = 0; q < 4; ++q) {
                v4f lv = lrow[q];     // broadcast (2 addrs/wave)
                v4f av = arow[q];     // conflict-free (padded stride)
                #pragma unroll
                for (int e = 0; e < 4; ++e) {
                    float raw = lpv + lv[e];
                    float lr  = raw > 0.0f ? raw : 0.2f * raw;
                    float lg  = (av[e] == 1.0f) ? lr : NEG_BIG;
                    pl[q * 4 + e] = lg;
                    tmax = fmaxf(tmax, lg);
                }
            }
            const float mnew  = fmaxf(m, tmax);
            const float scale = __expf(m - mnew);   // iter 0: exp(-inf)=0
            m = mnew;
            float psum = 0.0f;
            v16h bfrag;                              // P row of this lane
            #pragma unroll
            for (int j = 0; j < 16; ++j) {
                float e = __expf(pl[j] - mnew);
                psum += e;
                bfrag[j] = (_Float16)e;
            }
            s = s * scale + psum;
            #pragma unroll
            for (int r = 0; r < 8; ++r) acc[r] *= scale;

            // A: block-diagonal V^T halves; conflict-free b128 LDS read
            v8h hv0 = *(const v8h*)(hcur + (hp * 16 + l15) * HSTRIDE
                                        + js * 16);
            v8h hv1 = *(const v8h*)(hcur + (hp * 16 + l15) * HSTRIDE
                                        + js * 16 + 8);
            const bool on = ((l15 < 8) == (lh == 0));
            v16h afrag;
            #pragma unroll
            for (int e = 0; e < 8; ++e) {
                afrag[e]     = on ? hv0[e] : (_Float16)0.0f;
                afrag[e + 8] = on ? hv1[e] : (_Float16)0.0f;
            }
            acc = __builtin_amdgcn_wmma_f32_16x16x32_f16(
                false, afrag, false, bfrag, (short)0, acc, false, false);
        }
        wait_async();       // own async LDS writes for next tile complete
        __syncthreads();    // whole block done writing next / reading cur
    }

    const float inv = 1.0f / s;
    float* op = out + (size_t)nf * CC + head * 8;
    v4f o0, o1;
    #pragma unroll
    for (int r = 0; r < 4; ++r) { o0[r] = acc[r] * inv; o1[r] = acc[r + 4] * inv; }
    *(v4f*)op       = o0;
    *((v4f*)op + 1) = o1;
}

// ===========================================================================
extern "C" void kernel_launch(void* const* d_in, const int* in_sizes, int n_in,
                              void* d_out, int out_size, void* d_ws, size_t ws_size,
                              hipStream_t stream) {
    const float* X    = (const float*)d_in[0];
    const float* adj  = (const float*)d_in[1];
    const float* W    = (const float*)d_in[2];
    const float* bias = (const float*)d_in[3];
    const float* aw   = (const float*)d_in[4];
    float* out        = (float*)d_out;

    _Float16* hT  = (_Float16*)d_ws;                                   // 1 MB
    float*    lpT = (float*)((char*)d_ws + (1u << 20));                // 256 KB
    float*    lcT = (float*)((char*)d_ws + (1u << 20) + (256u << 10)); // 256 KB

    gat_proj_kernel<<<256, 256, 0, stream>>>(X, W, bias, aw, hT, lpT, lcT);
    gat_attn_kernel<<<BB * (NN / 16), 256, 0, stream>>>(adj, hT, lpT, lcT, out);

    // tuple output: (out, adj_matrix) concatenated flat
    (void)hipMemcpyAsync(out + (size_t)BB * NN * CC, adj,
                         sizeof(float) * (size_t)BB * NN * NN,
                         hipMemcpyDeviceToDevice, stream);
}
